// UpdateNode_14190571946519
// MI455X (gfx1250) — matrix operand order repaired and checked by reading, not compiled, not run on hardware
//
#include <hip/hip_runtime.h>
#include <hip/hip_bf16.h>

// ---------- constants from the reference ----------
#define NN 20000
#define EE 160000
#define FEAT 240
#define LAT 128

// c_old = 1/sqrt(1.25), c_new = 0.5*c_old, agg scale 1/sqrt(8)
#define C_OLD 0.8944271909999159f
#define C_NEW_S 0.15811388300841897f   // c_new / sqrt(8)
#define INV_SQRT_NT 0.10259783520851541f // 1/sqrt(95)

// workspace layout (in f32 words)
#define WS_AGG_OFF 0
#define WS_W_OFF   (NN * FEAT)                 // 4,800,000
// bf16 weight fragment segment offsets (in dwords, relative to WS_W_OFF)
#define OFF_WR   0        // 128x160 : 10 ntiles * 4 kchunks
#define OFF_WENV 10240    // 128x112 : 7*4
#define OFF_W0   17408    // 128x112 : 7*4
#define OFF_W1   24576    // 64x32   : 2*2
#define OFF_W2   25600    // 32x16   : 1*1
#define OFF_P0   25856    // 64x64   : 4*2
#define OFF_P1   27904    // 32x32   : 2*1
#define OFF_P2   28416    // 16(pad32)x16 : 1*1
#define WFRAG_DWORDS 28672

typedef __attribute__((ext_vector_type(16))) __bf16 v16bf;
typedef __attribute__((ext_vector_type(8)))  float  v8f;
typedef __attribute__((ext_vector_type(8)))  unsigned int v8u;

union BFrag { v8u u; v16bf b; uint4 q[2]; };

// f32 -> bf16 (RNE) pack of two values into one dword (low = first).
// v_fma_mixlo_bf16 / v_fma_mixhi_bf16 are documented CDNA5 VOP3P ops
// (ISA 15.10 opcodes 62/63): D[15:0] (resp [31:16]) = bf16(fma(s0,s1,s2)).
// fma(x, 1.0, 0) == x, so this is a 2-op RNE pack.
__device__ __forceinline__ unsigned int pk2(float a, float b) {
  unsigned int d;
  asm("v_fma_mixlo_bf16 %0, %1, 1.0, 0" : "=v"(d) : "v"(a));
  asm("v_fma_mixhi_bf16 %0, %1, 1.0, 0" : "+v"(d) : "v"(b));
  return d;
}

// fast sigmoid: v_exp + v_rcp
__device__ __forceinline__ float fastsig(float x) {
  return __builtin_amdgcn_rcpf(1.f + __expf(-x));
}

// A fragment (16x32 bf16) from a contiguous f32 row: rp points at k=0 of this
// lane's row. Layout (ISA 7.12.2): lanes<16 hold K 2v,2v+1 (v=0..3) and
// 16+2v (v=4..7); lanes>=16 the +8 variants.
__device__ __forceinline__ v16bf aFragContig(const float* rp, int hi) {
  BFrag fr;
  const float* p0 = rp + hi * 8;
  const float* p1 = rp + 16 + hi * 8;
#pragma unroll
  for (int v = 0; v < 4; ++v) {
    fr.u[v]     = pk2(p0[2 * v], p0[2 * v + 1]);
    fr.u[v + 4] = pk2(p1[2 * v], p1[2 * v + 1]);
  }
  return fr.b;
}

// A fragment with strided k (element k at rp[k*stride])
__device__ __forceinline__ v16bf aFragStride(const float* rp, int hi, int stride) {
  BFrag fr;
  int k0 = hi * 8;
#pragma unroll
  for (int v = 0; v < 4; ++v) {
    fr.u[v]     = pk2(rp[(k0 + 2 * v) * stride],      rp[(k0 + 2 * v + 1) * stride]);
    fr.u[v + 4] = pk2(rp[(16 + k0 + 2 * v) * stride], rp[(16 + k0 + 2 * v + 1) * stride]);
  }
  return fr.b;
}

// B fragment (32x16 bf16) from pre-swizzled weight storage: per fragment,
// dword index = lane*8 + v  (two coalesced uint4 loads per lane)
__device__ __forceinline__ v16bf bFrag(const unsigned int* wseg, int frag, int lane) {
  BFrag fr;
  const uint4* p = (const uint4*)(wseg + frag * 256 + lane * 8);
  fr.q[0] = p[0];
  fr.q[1] = p[1];
  return fr.b;
}

__device__ __forceinline__ v8f wmma_bf16(v16bf a, v16bf b, v8f c) {
  return __builtin_amdgcn_wmma_f32_16x16x32_bf16(false, a, false, b, (short)0, c,
                                                 false, false);
}

// one 16x16 C tile accumulated over 4 K-chunks; all 8 B-fragment b128 loads
// are issued before the first WMMA so they overlap the matrix pipe.
__device__ __forceinline__ v8f gemmTile4(const v16bf* a, const unsigned int* wseg,
                                         int tbase, int lane) {
  v16bf b0 = bFrag(wseg, tbase + 0, lane);
  v16bf b1 = bFrag(wseg, tbase + 1, lane);
  v16bf b2 = bFrag(wseg, tbase + 2, lane);
  v16bf b3 = bFrag(wseg, tbase + 3, lane);
  v8f c = {};
  c = wmma_bf16(a[0], b0, c);
  c = wmma_bf16(a[1], b1, c);
  c = wmma_bf16(a[2], b2, c);
  c = wmma_bf16(a[3], b3, c);
  return c;
}

__device__ __forceinline__ v8f gemmTile2(v16bf a0, v16bf a1, const unsigned int* wseg,
                                         int tbase, int lane) {
  v16bf b0 = bFrag(wseg, tbase + 0, lane);
  v16bf b1 = bFrag(wseg, tbase + 1, lane);
  v8f c = {};
  c = wmma_bf16(a0, b0, c);
  c = wmma_bf16(a1, b1, c);
  return c;
}

// envelope tile t: (latents @ Wenv + benv), computed lazily at point of use
__device__ __forceinline__ v8f envTile(const v16bf* aLat, const unsigned int* wfrag,
                                       const float* benv, int t, int lane, int lm) {
  v8f c = gemmTile4(aLat, wfrag + OFF_WENV, t * 4, lane);
  float bias = benv[t * 16 + lm];
#pragma unroll
  for (int r = 0; r < 8; ++r) c[r] += bias;
  return c;
}

__device__ __forceinline__ void atomAdd(float* p, float v) {
  (void)__hip_atomic_fetch_add(p, v, __ATOMIC_RELAXED, __HIP_MEMORY_SCOPE_AGENT);
}

// ---------------- tiny prep kernels ----------------
__global__ void zero_kernel(float* __restrict__ p, int n) {
  int i = blockIdx.x * blockDim.x + threadIdx.x;
  if (i < n) p[i] = 0.f;
}

// convert shared weights to bf16 in WMMA B-fragment layout
__global__ void wconv_kernel(const float* __restrict__ Wr, const float* __restrict__ Wenv,
                             const float* __restrict__ W0, const float* __restrict__ W1,
                             const float* __restrict__ W2, const float* __restrict__ P0,
                             const float* __restrict__ P1, const float* __restrict__ P2,
                             unsigned int* __restrict__ dst) {
  int i = blockIdx.x * blockDim.x + threadIdx.x;
  if (i >= WFRAG_DWORDS) return;
  const float* W; int Kr, Nc, nKc, base;
  if      (i < OFF_WENV) { W = Wr;   Kr = 128; Nc = 160; nKc = 4; base = OFF_WR; }
  else if (i < OFF_W0)   { W = Wenv; Kr = 128; Nc = 112; nKc = 4; base = OFF_WENV; }
  else if (i < OFF_W1)   { W = W0;   Kr = 128; Nc = 112; nKc = 4; base = OFF_W0; }
  else if (i < OFF_W2)   { W = W1;   Kr = 64;  Nc = 32;  nKc = 2; base = OFF_W1; }
  else if (i < OFF_P0)   { W = W2;   Kr = 32;  Nc = 16;  nKc = 1; base = OFF_W2; }
  else if (i < OFF_P1)   { W = P0;   Kr = 64;  Nc = 64;  nKc = 2; base = OFF_P0; }
  else if (i < OFF_P2)   { W = P1;   Kr = 32;  Nc = 32;  nKc = 1; base = OFF_P1; }
  else                   { W = P2;   Kr = 16;  Nc = 16;  nKc = 1; base = OFF_P2; }
  int d = i - base;
  int f = d >> 8, rem = d & 255, l = rem >> 3, v = rem & 7;
  int nt = f / nKc, kc = f % nKc;
  int ncol = (l & 15) + nt * 16;
  int k0 = kc * 32 + ((v >> 2) << 4) + ((l >> 4) << 3) + 2 * (v & 3);   // B layout: lanes>=16 hold K+8; v>=4 hold K+16
  float a = (k0 < Kr)     ? W[k0 * Nc + ncol]       : 0.f;
  float b = (k0 + 1 < Kr) ? W[(k0 + 1) * Nc + ncol] : 0.f;
  dst[i] = pk2(a, b);
}

// fused node update: find exact one-hot type index, gather per-type tune
// column directly (O rows are L2-resident), apply residual mix.
__global__ __launch_bounds__(256) void node_kernel(
    const float* __restrict__ nf, const float* __restrict__ agg,
    const float* __restrict__ oh, const float* __restrict__ O0,
    const float* __restrict__ O1, const float* __restrict__ O2,
    float* __restrict__ out) {
  __shared__ int sT;
  const int n = blockIdx.x;
  const int c = threadIdx.x;
  const float* ohr = oh + (size_t)n * 95;
  if (c < 95 && ohr[c] == 1.0f) sT = c;   // jax.nn.one_hot -> exact 1.0f
  __syncthreads();
  const int t = sT;
  if (c >= FEAT) return;
  float sval;
  if (c < 64)        sval = O0[c * 95 + t];
  else if (c < 160)  sval = O1[((c - 64) / 3) * 95 + t];
  else               sval = O2[((c - 160) / 5) * 95 + t];
  sval *= INV_SQRT_NT;
  size_t i = (size_t)n * FEAT + c;
  out[i] = (C_OLD * nf[i] + C_NEW_S * agg[i]) * (1.f + sval);
}

// ---------------- main fused edge kernel ----------------
// one wave (32 lanes) per 16-edge tile; all channel GEMMs via bf16 WMMA with
// f32 accumulation; per-edge rotations folded into Gram matrices G = D*D^T.
// rad0 tiles are fused into the y0 loop and env tiles are computed at point of
// use, keeping live accumulators to <= 3 tiles at any time.
__global__ __launch_bounds__(32) void edge_kernel(
    const float* __restrict__ latents, const float* __restrict__ nodef,
    const float* __restrict__ edgef, const float* __restrict__ D1,
    const float* __restrict__ D2, const float* __restrict__ benv,
    const float* __restrict__ b0, const int* __restrict__ eidx,
    const unsigned int* __restrict__ wfrag, float* __restrict__ agg) {
  __shared__ int   sCent[16];
  __shared__ float sG1[16][9];
  __shared__ float sG2[16][25];
  __shared__ float sF0[16][64];      // silu(y0[:, :64])
  __shared__ float sZ1[3][16][32];   // gated/scaled q1 per m
  __shared__ float sZ2[5][16][32];   // gated/scaled q2 per m (K padded to 32)

  const int lane = threadIdx.x;
  const int hi = lane >> 4;
  const int lm = lane & 15;
  const int ebase = blockIdx.x * 16;

  // warm L2/L0 for this tile's streamed rows
  __builtin_prefetch(edgef + (size_t)(ebase + lm) * FEAT + hi * 120, 0, 3);
  __builtin_prefetch(latents + (size_t)(ebase + lm) * LAT + hi * 64, 0, 3);

  // ---- phase 0: centers + Gram matrices G = D*D^T ----
  if (lane < 16) {
    sCent[lane] = eidx[ebase + lane];
    const float* d1 = D1 + (size_t)(ebase + lane) * 9;
    float d[9];
#pragma unroll
    for (int i = 0; i < 9; ++i) d[i] = d1[i];
#pragma unroll
    for (int a = 0; a < 3; ++a)
#pragma unroll
      for (int b = 0; b < 3; ++b)
        sG1[lane][a * 3 + b] =
            d[a * 3] * d[b * 3] + d[a * 3 + 1] * d[b * 3 + 1] + d[a * 3 + 2] * d[b * 3 + 2];
  } else {
    int e2 = lane - 16;
    const float* d2 = D2 + (size_t)(ebase + e2) * 25;
    float d[25];
#pragma unroll
    for (int i = 0; i < 25; ++i) d[i] = d2[i];
#pragma unroll
    for (int a = 0; a < 5; ++a)
#pragma unroll
      for (int b = 0; b < 5; ++b) {
        float sacc = 0.f;
#pragma unroll
        for (int m = 0; m < 5; ++m) sacc += d[a * 5 + m] * d[b * 5 + m];
        sG2[e2][a * 5 + b] = sacc;
      }
  }
  // zero the padded K half of sZ2
  for (int i = lane; i < 5 * 16 * 16; i += 32) {
    int m = i >> 8, r = (i >> 4) & 15, k = i & 15;
    sZ2[m][r][16 + k] = 0.f;
  }
  __syncthreads();

  // ---- A fragments: latents (kept resident) and x0 = [n0 | e0] ----
  const float* latRow = latents + (size_t)(ebase + lm) * LAT;
  v16bf aLat[4];
#pragma unroll
  for (int kc = 0; kc < 4; ++kc) aLat[kc] = aFragContig(latRow + kc * 32, hi);

  const int nodeA = sCent[lm];
  const float* nfRow = nodef + (size_t)nodeA * FEAT;
  const float* efRow = edgef + (size_t)(ebase + lm) * FEAT;
  v16bf aX0[4];
  aX0[0] = aFragContig(nfRow, hi);
  aX0[1] = aFragContig(nfRow + 32, hi);
  aX0[2] = aFragContig(efRow, hi);
  aX0[3] = aFragContig(efRow + 32, hi);

  // ---- fused loop: rad0 tile t + y0 tile t -> activations ----
  v8f gate[3];
#pragma unroll
  for (int t = 0; t < 7; ++t) {
    v8f r = gemmTile4(aLat, wfrag + OFF_WR, t * 4, lane);   // rad tile t
    v8f c = gemmTile4(aX0, wfrag + OFF_W0, t * 4, lane);    // y0 tile t
#pragma unroll
    for (int rr = 0; rr < 8; ++rr) c[rr] *= r[rr];
    if (t < 4) {
#pragma unroll
      for (int rr = 0; rr < 8; ++rr) {
        float x = c[rr];
        sF0[rr + hi * 8][t * 16 + lm] = x * fastsig(x);  // silu
      }
    } else {
#pragma unroll
      for (int rr = 0; rr < 8; ++rr) c[rr] = fastsig(c[rr]);  // sigmoid
      gate[t - 4] = c;
    }
  }
  // rad tiles 7..9 (rad1, rad2) kept resident
  v8f radHi[3];
#pragma unroll
  for (int tt = 0; tt < 3; ++tt)
    radHi[tt] = gemmTile4(aLat, wfrag + OFF_WR, (7 + tt) * 4, lane);

  // ---- l=1: q1[m] = x1[:,:,m] @ W1, gated+scaled -> sZ1 ----
#pragma unroll
  for (int m = 0; m < 3; ++m) {
    v16bf a0 = aFragStride(nfRow + 64 + m, hi, 3);  // c = 0..31 (node)
    v16bf a1 = aFragStride(efRow + 64 + m, hi, 3);  // c = 32..63 (edge)
#pragma unroll
    for (int t = 0; t < 2; ++t) {
      v8f c = gemmTile2(a0, a1, wfrag + OFF_W1, t * 2, lane);
#pragma unroll
      for (int r = 0; r < 8; ++r)
        sZ1[m][r + hi * 8][t * 16 + lm] = c[r] * radHi[t][r] * gate[t][r];
    }
  }

  // ---- l=2: q2[m] = x2[:,:,m] @ W2, gated+scaled -> sZ2 ----
#pragma unroll
  for (int m = 0; m < 5; ++m) {
    BFrag fr;
    const float* pn = nfRow + 160 + m;
    const float* pe = efRow + 160 + m;
    int k0 = hi * 8;
#pragma unroll
    for (int v = 0; v < 4; ++v) {
      fr.u[v]     = pk2(pn[(k0 + 2 * v) * 5], pn[(k0 + 2 * v + 1) * 5]);  // c 0..15 node
      fr.u[v + 4] = pk2(pe[(k0 + 2 * v) * 5], pe[(k0 + 2 * v + 1) * 5]);  // c 16..31 edge
    }
    v16bf bW2 = bFrag(wfrag + OFF_W2, 0, lane);
    v8f c = {};
    c = wmma_bf16(fr.b, bW2, c);
#pragma unroll
    for (int r = 0; r < 8; ++r)
      sZ2[m][r + hi * 8][lm] = c[r] * radHi[2][r] * gate[2][r];
  }
  __syncthreads();

  // centers in C-fragment row order
  int centC[8];
#pragma unroll
  for (int r = 0; r < 8; ++r) centC[r] = sCent[r + hi * 8];

  // ---- p0 = silu(y0[:, :64]) @ P0 + b0 ; m0 = p0 * env[:, :64] ----
  v16bf aF0 = aFragContig(&sF0[lm][0], hi);
  v16bf aF1 = aFragContig(&sF0[lm][0] + 32, hi);
#pragma unroll
  for (int t = 0; t < 4; ++t) {
    v8f c = gemmTile2(aF0, aF1, wfrag + OFF_P0, t * 2, lane);
    v8f e = envTile(aLat, wfrag, benv, t, lane, lm);
    float bias = b0[t * 16 + lm];
#pragma unroll
    for (int r = 0; r < 8; ++r)
      atomAdd(agg + (size_t)centC[r] * FEAT + t * 16 + lm, (c[r] + bias) * e[r]);
  }

  // ---- p1 = z1 @ P1, rotate by G1, envelope, scatter ----
  v8f p1pre[3][2];
#pragma unroll
  for (int m = 0; m < 3; ++m) {
    v16bf aZ = aFragContig(&sZ1[m][lm][0], hi);
#pragma unroll
    for (int t = 0; t < 2; ++t) {
      v16bf b = bFrag(wfrag + OFF_P1, t, lane);
      v8f c = {};
      p1pre[m][t] = wmma_bf16(aZ, b, c);
    }
  }
#pragma unroll
  for (int t = 0; t < 2; ++t) {
    v8f e = envTile(aLat, wfrag, benv, 4 + t, lane, lm);
#pragma unroll
    for (int n = 0; n < 3; ++n)
#pragma unroll
      for (int r = 0; r < 8; ++r) {
        int ee = r + hi * 8;
        float v = p1pre[0][t][r] * sG1[ee][n] + p1pre[1][t][r] * sG1[ee][3 + n] +
                  p1pre[2][t][r] * sG1[ee][6 + n];
        v *= e[r];
        atomAdd(agg + (size_t)centC[r] * FEAT + 64 + (t * 16 + lm) * 3 + n, v);
      }
  }

  // ---- p2 = z2 @ P2, rotate by G2, envelope, scatter ----
  v8f p2pre[5];
#pragma unroll
  for (int m = 0; m < 5; ++m) {
    v16bf aZ = aFragContig(&sZ2[m][lm][0], hi);
    v16bf b = bFrag(wfrag + OFF_P2, 0, lane);
    v8f c = {};
    p2pre[m] = wmma_bf16(aZ, b, c);
  }
  {
    v8f e = envTile(aLat, wfrag, benv, 6, lane, lm);
#pragma unroll
    for (int n = 0; n < 5; ++n)
#pragma unroll
      for (int r = 0; r < 8; ++r) {
        int ee = r + hi * 8;
        float v = 0.f;
#pragma unroll
        for (int m = 0; m < 5; ++m) v += p2pre[m][r] * sG2[ee][m * 5 + n];
        v *= e[r];
        atomAdd(agg + (size_t)centC[r] * FEAT + 160 + lm * 5 + n, v);
      }
  }
}

// ---------------- launcher ----------------
extern "C" void kernel_launch(void* const* d_in, const int* in_sizes, int n_in,
                              void* d_out, int out_size, void* d_ws, size_t ws_size,
                              hipStream_t stream) {
  (void)in_sizes; (void)n_in; (void)out_size; (void)ws_size;
  const float* latents = (const float*)d_in[0];
  const float* nodef   = (const float*)d_in[1];
  const float* edgef   = (const float*)d_in[2];
  const float* onehot  = (const float*)d_in[3];
  const float* D1      = (const float*)d_in[4];
  const float* D2      = (const float*)d_in[5];
  const float* W0      = (const float*)d_in[6];
  const float* W1      = (const float*)d_in[7];
  const float* W2      = (const float*)d_in[8];
  const float* Wr      = (const float*)d_in[9];
  const float* P0      = (const float*)d_in[10];
  const float* b0      = (const float*)d_in[11];
  const float* P1      = (const float*)d_in[12];
  const float* P2      = (const float*)d_in[13];
  const float* Wenv    = (const float*)d_in[14];
  const float* benv    = (const float*)d_in[15];
  const float* O0      = (const float*)d_in[16];
  const float* O1      = (const float*)d_in[17];
  const float* O2      = (const float*)d_in[18];
  const int*   eidx    = (const int*)d_in[19];  // row 0 = edge_center
  float* out = (float*)d_out;
  float* ws  = (float*)d_ws;

  float* agg = ws + WS_AGG_OFF;
  unsigned int* wfrag = (unsigned int*)(ws + WS_W_OFF);

  // 1) zero the aggregation buffer (atomics accumulate into it)
  zero_kernel<<<(NN * FEAT + 255) / 256, 256, 0, stream>>>(agg, NN * FEAT);
  // 2) convert weights into bf16 WMMA B-fragment layout
  wconv_kernel<<<(WFRAG_DWORDS + 255) / 256, 256, 0, stream>>>(
      Wr, Wenv, W0, W1, W2, P0, P1, P2, wfrag);
  // 3) fused per-edge message kernel (WMMA) with atomic scatter
  edge_kernel<<<EE / 16, 32, 0, stream>>>(latents, nodef, edgef, D1, D2, benv, b0,
                                          eidx, wfrag, agg);
  // 4) fused node update (type gather + residual mix)
  node_kernel<<<NN, 256, 0, stream>>>(nodef, agg, onehot, O0, O1, O2, out);
}
